// MODEL3_14920716387045
// MI455X (gfx1250) — matrix operand order repaired and checked by reading, not compiled, not run on hardware
//
#include <hip/hip_runtime.h>
#include <math.h>

// ---------------- problem constants ----------------
#define LNUM   2
#define HNUM   8
#define DHEAD  64
#define DM     512
#define FF     1024
#define BSZ    8
#define SEQ    4096
#define NTOK   (BSZ * SEQ)       // 32768
#define FIN    586
#define EPSA   1e-6f
#define EPSLN  1e-5f
#define NCH    8                 // S-chunks for kv partials (deterministic)

typedef __attribute__((ext_vector_type(16))) _Float16 v16h;
typedef __attribute__((ext_vector_type(8)))  _Float16 v8h;
typedef __attribute__((ext_vector_type(8)))  float    v8f;
typedef __attribute__((ext_vector_type(4)))  unsigned v4u;
typedef __attribute__((ext_vector_type(8)))  int      v8i;
typedef __attribute__((ext_vector_type(4)))  int      v4i;

// ---------------- WMMA helpers (wave32, 16x16x32 f16 -> f32) ----------------
__device__ __forceinline__ v8f wmma_f16(v16h a, v16h b, v8f c) {
    return __builtin_amdgcn_wmma_f32_16x16x32_f16(
        /*neg_a=*/false, a, /*neg_b=*/false, b,
        /*c_mod=*/(short)0, c, /*reuse_a=*/false, /*reuse_b=*/false);
}

// A operand: LDS tile row-major [M][ldk] f16.  ISA 16-bit A 16x32 layout:
// lanes 0-15 -> K 0..7 & 16..23, lanes 16-31 -> K 8..15 & 24..31, row = M = lane&15
__device__ __forceinline__ v16h frag_a(const _Float16* lds, int mrow, int ldk,
                                       int kb, int lane) {
    int hf = (lane >> 4) & 1;
    int r  = lane & 15;
    const _Float16* p = lds + (size_t)(mrow + r) * ldk + kb + hf * 8;
    v8h lo = *(const v8h*)p;
    v8h hi = *(const v8h*)(p + 16);
    return __builtin_shufflevector(lo, hi, 0,1,2,3,4,5,6,7,8,9,10,11,12,13,14,15);
}

// B operand: LDS tile holds B transposed: [N][ldk] f16 (each row = one output col).
// ISA 16-bit B 32x16 layout: lanes 0-15 -> K 0..7 & 16..23, lanes 16-31 -> K 8..15 & 24..31, col = lane&15
__device__ __forceinline__ v16h frag_b(const _Float16* lds, int ncol, int ldk,
                                       int kb, int lane) {
    int hf = (lane >> 4) & 1;
    int r  = lane & 15;
    const _Float16* p = lds + (size_t)(ncol + r) * ldk + kb + hf * 8;
    v8h lo = *(const v8h*)p;
    v8h hi = *(const v8h*)(p + 16);
    return __builtin_shufflevector(lo, hi, 0,1,2,3,4,5,6,7,8,9,10,11,12,13,14,15);
}

// ---------------- TDM: DMA a [tile_n=128][tile_k=32] f16 tile of the pre-transposed
// weight Wt[N][K] into LDS with 16B row padding (row stride 80B == LDK 40 f16).
// Descriptor layout per CDNA5 ISA; dims are relative to
// tile start so the K tail (e.g. K=586) is zero-filled by TDM OOB handling.
// This toolchain exposes the 6-arg builtin:
//   (uint32x4 g0, int32x8 g1, int32x4 g2, int32x4 g3, int32x8 g4, i32 cpol)
__device__ __forceinline__ void tdm_load_b_tile(unsigned lds_off,
                                                const _Float16* wt,
                                                int row0, int kb, int K, int Nc) {
    unsigned long long ga =
        (unsigned long long)(size_t)(wt + (size_t)row0 * K + kb);
    v4u g0;
    g0[0] = 1u;                                    // count=1, is_restore=0, no gather
    g0[1] = lds_off;                               // LDS byte offset
    g0[2] = (unsigned)ga;                          // global_addr[31:0]
    g0[3] = (unsigned)(ga >> 32) | (2u << 30);     // global_addr[56:32] | type=2
    unsigned td0 = (unsigned)(K - kb);             // remaining K (OOB zero-fill)
    unsigned td1 = (unsigned)(Nc - row0);          // remaining rows
    v8i g1;
    g1[0] = (int)((1u << 16)      // data_size = 1 -> 2 bytes
                | (1u << 20)      // pad_enable
                | (3u << 22)      // pad_interval: 16 DWORDs (64B) between pads
                | (3u << 25));    // pad_amount: 4 DWORDs (16B)
    g1[1] = (int)((td0 & 0xFFFFu) << 16);                    // tensor_dim0 lo16
    g1[2] = (int)((td0 >> 16) | ((td1 & 0xFFFFu) << 16));    // dim0 hi / dim1 lo
    g1[3] = (int)((td1 >> 16) | (32u << 16));                // dim1 hi / tile_dim0=32
    g1[4] = 128;                                             // tile_dim1=128, tile_dim2=0
    g1[5] = K;                                               // tensor_dim0_stride
    g1[6] = 0;
    g1[7] = 0;
    v4i gz4 = {0, 0, 0, 0};                                  // 2D: groups 2/3 unused
    v8i gz8 = {0, 0, 0, 0, 0, 0, 0, 0};
    __builtin_amdgcn_tensor_load_to_lds(g0, g1, gz4, gz4, gz8, 0);
}

// ---------------- generic GEMM: out[M,Nc] = epi(A[M,K] @ W[K,Nc] + bias) ----------------
// A: f32 [M,K]; Wt: f16 pre-transposed [Nc,K] (TDM-loaded tiles).
// EPI: 0 = +bias ; 1 = feat (elu+1) ; 2 = relu ; 3 = +bias + residual
// grid = (Nc/128, Mrows/128), block = 256 (8 waves, each 32x64)
template <int EPI>
__global__ __launch_bounds__(256) void gemm_f16_kernel(
    const float* __restrict__ A, int lda,
    const _Float16* __restrict__ Wt,
    const float* __restrict__ bias,
    const float* __restrict__ res,
    float* __restrict__ out,
    int K, int Nc)
{
    const int LDK = 40;                       // 32 + 8 pad, 80B rows (16B aligned)
    __shared__ alignas(16) _Float16 As[128 * LDK];
    __shared__ alignas(16) _Float16 Bs[128 * LDK];

    const int t    = threadIdx.x;
    const int wave = t >> 5;
    const int lane = t & 31;
    const int m0   = blockIdx.y * 128;
    const int n0   = blockIdx.x * 128;
    const int wm   = (wave >> 1) * 32;        // 0,32,64,96
    const int wn   = (wave & 1) * 64;         // 0,64
    const unsigned bs_off = (unsigned)(size_t)(&Bs[0]);   // LDS byte offset

    v8f acc[8];
    for (int i = 0; i < 8; ++i) acc[i] = v8f{0.f,0.f,0.f,0.f,0.f,0.f,0.f,0.f};

    const int ksteps = (K + 31) >> 5;
    for (int ks = 0; ks < ksteps; ++ks) {
        const int kb = ks << 5;
        // B tile via Tensor Data Mover (wave 0 issues, waits, barrier publishes)
        if (t < 32) {
            tdm_load_b_tile(bs_off, Wt, n0, kb, K, Nc);
            __builtin_amdgcn_s_wait_tensorcnt(0);
        }
        // A tile: 128x32 f32 -> f16 (zero-pad K remainder)
        #pragma unroll
        for (int i = 0; i < 16; ++i) {
            int e = t + i * 256;
            int r = e >> 5, c = e & 31;
            int gk = kb + c;
            float v = (gk < K) ? A[(size_t)(m0 + r) * lda + gk] : 0.f;
            As[r * LDK + c] = (_Float16)v;
        }
        __syncthreads();

        v16h af[2], bf[4];
        af[0] = frag_a(As, wm,      LDK, 0, lane);
        af[1] = frag_a(As, wm + 16, LDK, 0, lane);
        #pragma unroll
        for (int j = 0; j < 4; ++j) bf[j] = frag_b(Bs, wn + j * 16, LDK, 0, lane);
        #pragma unroll
        for (int mt = 0; mt < 2; ++mt)
            #pragma unroll
            for (int nt = 0; nt < 4; ++nt)
                acc[mt * 4 + nt] = wmma_f16(af[mt], bf[nt], acc[mt * 4 + nt]);
        __syncthreads();
    }

    // epilogue: lane = N (lane&15), vgpr e = M (e + 8*(lane>=16))
    const int hf = lane >> 4, r = lane & 15;
    #pragma unroll
    for (int mt = 0; mt < 2; ++mt)
        #pragma unroll
        for (int nt = 0; nt < 4; ++nt) {
            v8f a = acc[mt * 4 + nt];
            int gn = n0 + wn + nt * 16 + r;
            float bv = bias ? bias[gn] : 0.f;
            #pragma unroll
            for (int e = 0; e < 8; ++e) {
                int gm = m0 + wm + mt * 16 + hf * 8 + e;
                float v = a[e] + bv;
                if (EPI == 1) v = (v > 0.f) ? (v + 1.f) : __expf(v);   // elu(x)+1
                if (EPI == 2) v = fmaxf(v, 0.f);
                if (EPI == 3) v += res[(size_t)gm * Nc + gn];
                out[(size_t)gm * Nc + gn] = v;
            }
        }
}

// ---------------- weight pre-transpose: Wt[n*K+k] = (f16) W[k*Nc+n] ----------------
__global__ __launch_bounds__(256) void wt_kernel(
    const float* __restrict__ W, _Float16* __restrict__ Wt, int K, int Nc)
{
    int i = blockIdx.x * 256 + threadIdx.x;
    if (i >= K * Nc) return;
    int k = i / Nc, n = i % Nc;                // coalesced reads
    Wt[(size_t)n * K + k] = (_Float16)W[i];
}

// ---------------- kv partials: kvpart[bh][ch][d][m] = sum_{s in chunk} k[s,d] v[s,m] ----------------
// grid = (64, NCH), block = 128 (4 waves, each 32x32)
__global__ __launch_bounds__(128) void kv_part_kernel(
    const float* __restrict__ kbuf, const float* __restrict__ vbuf,
    float* __restrict__ kvpart)
{
    const int LDK = 40;
    __shared__ alignas(16) _Float16 kT[64 * LDK];   // [d][s] (A, rows = d)
    __shared__ alignas(16) _Float16 vT[64 * LDK];   // [m][s] (B transposed)

    const int bh = blockIdx.x;           // b*8 + h
    const int b  = bh >> 3, h = bh & 7;
    const int s0 = blockIdx.y * (SEQ / NCH);
    const int t = threadIdx.x, wave = t >> 5, lane = t & 31;
    const int wm = (wave >> 1) * 32, wn = (wave & 1) * 32;

    v8f acc[4];
    for (int i = 0; i < 4; ++i) acc[i] = v8f{0.f,0.f,0.f,0.f,0.f,0.f,0.f,0.f};

    for (int ks = 0; ks < (SEQ / NCH) / 32; ++ks) {
        const int sb = s0 + ks * 32;
        #pragma unroll
        for (int i = 0; i < 16; ++i) {           // 2048 elems per matrix
            int e = t + i * 128;
            int s = e >> 6, d = e & 63;
            size_t g = ((size_t)(b * SEQ + sb + s)) * DM + h * DHEAD + d;
            kT[d * LDK + s] = (_Float16)kbuf[g];
            vT[d * LDK + s] = (_Float16)vbuf[g];
        }
        __syncthreads();
        v16h af[2], bf[2];
        af[0] = frag_a(kT, wm,      LDK, 0, lane);
        af[1] = frag_a(kT, wm + 16, LDK, 0, lane);
        bf[0] = frag_b(vT, wn,      LDK, 0, lane);
        bf[1] = frag_b(vT, wn + 16, LDK, 0, lane);
        acc[0] = wmma_f16(af[0], bf[0], acc[0]);
        acc[1] = wmma_f16(af[0], bf[1], acc[1]);
        acc[2] = wmma_f16(af[1], bf[0], acc[2]);
        acc[3] = wmma_f16(af[1], bf[1], acc[3]);
        __syncthreads();
    }

    const int hf = lane >> 4, r = lane & 15;
    float* base = kvpart + ((size_t)bh * NCH + blockIdx.y) * (64 * 64);
    #pragma unroll
    for (int mt = 0; mt < 2; ++mt)
        #pragma unroll
        for (int nt = 0; nt < 2; ++nt) {
            v8f a = acc[mt * 2 + nt];
            #pragma unroll
            for (int e = 0; e < 8; ++e) {
                int d = wm + mt * 16 + hf * 8 + e;
                int m = wn + nt * 16 + r;
                base[d * 64 + m] = a[e];
            }
        }
}

__global__ __launch_bounds__(256) void kv_reduce_kernel(
    const float* __restrict__ part, float* __restrict__ kv)
{
    int i = blockIdx.x * 256 + threadIdx.x;     // 64 * 4096
    int bh = i >> 12, rm = i & 4095;
    float s = 0.f;
    #pragma unroll
    for (int ch = 0; ch < NCH; ++ch)
        s += part[((size_t)bh * NCH + ch) * 4096 + rm];
    kv[(size_t)bh * 4096 + rm] = s;
}

// ---------------- ksum[b][c] = sum_s k[b,s,c]   grid=(B, DM/64), block=256 ----------------
__global__ __launch_bounds__(256) void ksum_kernel(
    const float* __restrict__ kbuf, float* __restrict__ ksum)
{
    __shared__ float red[256];
    const int b = blockIdx.x, c0 = blockIdx.y * 64;
    const int t = threadIdx.x;
    const int c = c0 + (t & 63), part = t >> 6;          // 4 partials per col
    float s = 0.f;
    for (int si = part * (SEQ / 4); si < (part + 1) * (SEQ / 4); ++si)
        s += kbuf[((size_t)(b * SEQ + si)) * DM + c];
    red[t] = s;
    __syncthreads();
    if (part == 0)
        ksum[b * DM + c] = red[t] + red[t + 64] + red[t + 128] + red[t + 192];
}

// ---------------- z[n*8+h] = 1/(q[n,h,:] . ksum[b,h,:] + eps) ----------------
__global__ __launch_bounds__(256) void z_kernel(
    const float* __restrict__ q, const float* __restrict__ ksum,
    float* __restrict__ z)
{
    int i = blockIdx.x * 256 + threadIdx.x;      // NTOK * HNUM
    int n = i >> 3, h = i & 7;
    int b = n >> 12;                             // n / SEQ
    const float* qr = q + (size_t)n * DM + h * DHEAD;
    const float* ks = ksum + (b * HNUM + h) * DHEAD;
    float s = 0.f;
    #pragma unroll
    for (int d = 0; d < DHEAD; ++d) s += qr[d] * ks[d];
    z[i] = 1.f / (s + EPSA);
}

// ---------------- a[n, h*64+m] = z * (q[n,h,:] @ kv[bh])   grid=(64, SEQ/128), block=256 ----------------
__global__ __launch_bounds__(256) void attn_av_kernel(
    const float* __restrict__ q, const float* __restrict__ kv,
    const float* __restrict__ z, float* __restrict__ aout)
{
    const int LDK = 72;                           // 64 + 8 pad, 144B rows
    __shared__ alignas(16) _Float16 Qs[128 * LDK];
    __shared__ alignas(16) _Float16 Bs[64 * LDK];

    const int bh = blockIdx.x, b = bh >> 3, h = bh & 7;
    const int s0 = blockIdx.y * 128;
    const int t = threadIdx.x, wave = t >> 5, lane = t & 31;
    const int wm = (wave >> 1) * 32, wn = (wave & 1) * 32;

    #pragma unroll
    for (int i = 0; i < 32; ++i) {               // 128x64 q tile
        int e = t + i * 256;
        int r = e >> 6, kk = e & 63;
        Qs[r * LDK + kk] =
            (_Float16)q[((size_t)(b * SEQ + s0 + r)) * DM + h * DHEAD + kk];
    }
    #pragma unroll
    for (int i = 0; i < 16; ++i) {               // kv 64x64 transposed -> [m][d]
        int e = t + i * 256;
        int d = e >> 6, m = e & 63;
        Bs[m * LDK + d] = (_Float16)kv[(size_t)bh * 4096 + e];
    }
    __syncthreads();

    v8f acc[4];
    for (int i = 0; i < 4; ++i) acc[i] = v8f{0.f,0.f,0.f,0.f,0.f,0.f,0.f,0.f};
    #pragma unroll
    for (int kb = 0; kb < 64; kb += 32) {
        v16h af[2], bf[2];
        af[0] = frag_a(Qs, wm,      LDK, kb, lane);
        af[1] = frag_a(Qs, wm + 16, LDK, kb, lane);
        bf[0] = frag_b(Bs, wn,      LDK, kb, lane);
        bf[1] = frag_b(Bs, wn + 16, LDK, kb, lane);
        acc[0] = wmma_f16(af[0], bf[0], acc[0]);
        acc[1] = wmma_f16(af[0], bf[1], acc[1]);
        acc[2] = wmma_f16(af[1], bf[0], acc[2]);
        acc[3] = wmma_f16(af[1], bf[1], acc[3]);
    }

    const int hf = lane >> 4, r = lane & 15;
    #pragma unroll
    for (int mt = 0; mt < 2; ++mt)
        #pragma unroll
        for (int nt = 0; nt < 2; ++nt) {
            v8f a = acc[mt * 2 + nt];
            int m = wn + nt * 16 + r;
            #pragma unroll
            for (int e = 0; e < 8; ++e) {
                int tok = b * SEQ + s0 + wm + mt * 16 + hf * 8 + e;
                float zz = z[tok * HNUM + h];
                aout[(size_t)tok * DM + h * DHEAD + m] = a[e] * zz;
            }
        }
}

// ---------------- LayerNorm: wave-per-row (wave32), 8 rows/block ----------------
__global__ __launch_bounds__(256) void ln_kernel(
    const float* __restrict__ in, const float* __restrict__ sc,
    const float* __restrict__ bi, float* __restrict__ out)
{
    const int wave = threadIdx.x >> 5, lane = threadIdx.x & 31;
    const int row = blockIdx.x * 8 + wave;
    const float* x = in + (size_t)row * DM;
    float v[16];
    float s = 0.f;
    #pragma unroll
    for (int i = 0; i < 16; ++i) { v[i] = x[lane + 32 * i]; s += v[i]; }
    #pragma unroll
    for (int off = 16; off > 0; off >>= 1) s += __shfl_xor(s, off, 32);
    float mu = s * (1.f / DM);
    float s2 = 0.f;
    #pragma unroll
    for (int i = 0; i < 16; ++i) { float d = v[i] - mu; s2 += d * d; }
    #pragma unroll
    for (int off = 16; off > 0; off >>= 1) s2 += __shfl_xor(s2, off, 32);
    float rstd = rsqrtf(s2 * (1.f / DM) + EPSLN);
    #pragma unroll
    for (int i = 0; i < 16; ++i) {
        int c = lane + 32 * i;
        out[(size_t)row * DM + c] = (v[i] - mu) * rstd * sc[c] + bi[c];
    }
}

// ---------------- final projection: out[n] = x[n,:] . Wout + bout ----------------
__global__ __launch_bounds__(256) void outproj_kernel(
    const float* __restrict__ x, const float* __restrict__ Wout,
    const float* __restrict__ bout, float* __restrict__ out)
{
    const int wave = threadIdx.x >> 5, lane = threadIdx.x & 31;
    const int row = blockIdx.x * 8 + wave;
    float s = 0.f;
    #pragma unroll
    for (int i = 0; i < 16; ++i) {
        int c = lane + 32 * i;
        s += x[(size_t)row * DM + c] * Wout[c];
    }
    #pragma unroll
    for (int off = 16; off > 0; off >>= 1) s += __shfl_xor(s, off, 32);
    if (lane == 0) out[row] = s + bout[0];
}

// ---------------- host orchestration ----------------
extern "C" void kernel_launch(void* const* d_in, const int* in_sizes, int n_in,
                              void* d_out, int out_size, void* d_ws, size_t ws_size,
                              hipStream_t stream)
{
    (void)in_sizes; (void)n_in; (void)out_size; (void)ws_size;
    const float* emb   = (const float*)d_in[0];
    const float* W0    = (const float*)d_in[1];
    const float* b0    = (const float*)d_in[2];
    const float* Wq    = (const float*)d_in[3];
    const float* bq    = (const float*)d_in[4];
    const float* Wk    = (const float*)d_in[5];
    const float* bk    = (const float*)d_in[6];
    const float* Wv    = (const float*)d_in[7];
    const float* bv    = (const float*)d_in[8];
    const float* Wo    = (const float*)d_in[9];
    const float* bo    = (const float*)d_in[10];
    const float* ln1s  = (const float*)d_in[11];
    const float* ln1b  = (const float*)d_in[12];
    const float* W1    = (const float*)d_in[13];
    const float* b1    = (const float*)d_in[14];
    const float* W2    = (const float*)d_in[15];
    const float* b2    = (const float*)d_in[16];
    const float* ln2s  = (const float*)d_in[17];
    const float* ln2b  = (const float*)d_in[18];
    const float* lnfs  = (const float*)d_in[19];
    const float* lnfb  = (const float*)d_in[20];
    const float* Wout  = (const float*)d_in[21];
    const float* bout  = (const float*)d_in[22];

    // workspace partition (floats)
    float* ws = (float*)d_ws;
    const size_t SZ = (size_t)NTOK * DM;       // 16,777,216
    float* xbuf   = ws;                        // [N,512]
    float* qbuf   = xbuf + SZ;
    float* kbuf   = qbuf + SZ;                 // reused as 'a' after attention
    float* vbuf   = kbuf + SZ;                 // reused as x+attn, final ln
    float* ybuf   = vbuf + SZ;                 // [N,1024]
    float* kvpart = ybuf + (size_t)NTOK * FF;  // [64][NCH][64][64]
    float* kvbuf  = kvpart + (size_t)64 * NCH * 4096;
    float* ksum   = kvbuf + (size_t)64 * 4096;
    float* zbuf   = ksum + BSZ * DM;           // [N,8]
    // f16 pre-transposed weights
    _Float16* wt0 = (_Float16*)(zbuf + (size_t)NTOK * HNUM);   // [512][586]
    _Float16* wtl = wt0 + (size_t)DM * FIN;    // per-layer block of transposed weights
    const size_t PER_L = 4 * (size_t)DM * DM + 2 * (size_t)DM * FF;

    const dim3 blk256(256), blk128(128);
    const dim3 gProj(DM / 128, NTOK / 128);    // (4,256)
    const dim3 gFF1(FF / 128, NTOK / 128);     // (8,256)

    // ---- one-time weight transposes to f16 [N][K] (TDM source layout) ----
    wt_kernel<<<dim3((FIN * DM) / 256), blk256, 0, stream>>>(W0, wt0, FIN, DM);
    for (int l = 0; l < LNUM; ++l) {
        _Float16* base = wtl + (size_t)l * PER_L;
        wt_kernel<<<dim3(1024), blk256, 0, stream>>>(Wq + (size_t)l * DM * DM, base,                 DM, DM);
        wt_kernel<<<dim3(1024), blk256, 0, stream>>>(Wk + (size_t)l * DM * DM, base + 1 * DM * DM,   DM, DM);
        wt_kernel<<<dim3(1024), blk256, 0, stream>>>(Wv + (size_t)l * DM * DM, base + 2 * DM * DM,   DM, DM);
        wt_kernel<<<dim3(1024), blk256, 0, stream>>>(Wo + (size_t)l * DM * DM, base + 3 * DM * DM,   DM, DM);
        wt_kernel<<<dim3(2048), blk256, 0, stream>>>(W1 + (size_t)l * DM * FF, base + 4 * DM * DM,   DM, FF);
        wt_kernel<<<dim3(2048), blk256, 0, stream>>>(W2 + (size_t)l * FF * DM, base + 4 * DM * DM + DM * FF, FF, DM);
    }

    // x = emb @ W0 + b0
    gemm_f16_kernel<0><<<gProj, blk256, 0, stream>>>(emb, FIN, wt0, b0, nullptr,
                                                     xbuf, FIN, DM);
    for (int l = 0; l < LNUM; ++l) {
        _Float16* base = wtl + (size_t)l * PER_L;
        const _Float16* wtq = base;
        const _Float16* wtk = base + 1 * DM * DM;
        const _Float16* wtv = base + 2 * DM * DM;
        const _Float16* wto = base + 3 * DM * DM;
        const _Float16* wt1 = base + 4 * DM * DM;
        const _Float16* wt2 = base + 4 * DM * DM + DM * FF;

        gemm_f16_kernel<1><<<gProj, blk256, 0, stream>>>(xbuf, DM, wtq,
            bq + l * DM, nullptr, qbuf, DM, DM);                       // q = feat(xWq+bq)
        gemm_f16_kernel<1><<<gProj, blk256, 0, stream>>>(xbuf, DM, wtk,
            bk + l * DM, nullptr, kbuf, DM, DM);                       // k = feat(xWk+bk)
        gemm_f16_kernel<0><<<gProj, blk256, 0, stream>>>(xbuf, DM, wtv,
            bv + l * DM, nullptr, vbuf, DM, DM);                       // v = xWv+bv

        kv_part_kernel<<<dim3(BSZ * HNUM, NCH), blk128, 0, stream>>>(kbuf, vbuf, kvpart);
        kv_reduce_kernel<<<dim3(64 * 4096 / 256), blk256, 0, stream>>>(kvpart, kvbuf);
        ksum_kernel<<<dim3(BSZ, DM / 64), blk256, 0, stream>>>(kbuf, ksum);
        z_kernel<<<dim3(NTOK * HNUM / 256), blk256, 0, stream>>>(qbuf, ksum, zbuf);

        // a -> kbuf (k dead after kv/ksum)
        attn_av_kernel<<<dim3(BSZ * HNUM, SEQ / 128), blk256, 0, stream>>>(
            qbuf, kvbuf, zbuf, kbuf);

        // x_attn = x + a @ Wo + bo  -> vbuf (v dead)
        gemm_f16_kernel<3><<<gProj, blk256, 0, stream>>>(kbuf, DM, wto,
            bo + l * DM, xbuf, vbuf, DM, DM);
        ln_kernel<<<dim3(NTOK / 8), blk256, 0, stream>>>(vbuf,
            ln1s + l * DM, ln1b + l * DM, xbuf);

        // FFN
        gemm_f16_kernel<2><<<gFF1, blk256, 0, stream>>>(xbuf, DM, wt1,
            b1 + l * FF, nullptr, ybuf, DM, FF);                       // relu(xW1+b1)
        gemm_f16_kernel<3><<<gProj, blk256, 0, stream>>>(ybuf, FF, wt2,
            b2 + l * DM, xbuf, qbuf, FF, DM);                          // x + yW2+b2
        ln_kernel<<<dim3(NTOK / 8), blk256, 0, stream>>>(qbuf,
            ln2s + l * DM, ln2b + l * DM, xbuf);
    }
    // final LN + output projection
    ln_kernel<<<dim3(NTOK / 8), blk256, 0, stream>>>(xbuf, lnfs, lnfb, vbuf);
    outproj_kernel<<<dim3(NTOK / 8), blk256, 0, stream>>>(vbuf, Wout, bout,
                                                          (float*)d_out);
}
